// Net_24111946399802
// MI455X (gfx1250) — compile-verified, matrix operand order checked by reading, not run on hardware
//
#include <hip/hip_runtime.h>
#include <hip/hip_fp16.h>

typedef _Float16 h16;
typedef __attribute__((ext_vector_type(16))) _Float16 v16h;
typedef __attribute__((ext_vector_type(8)))  float    v8f;

#define NN 8192
#define LL 512
#define LDT 40  // padded LDS tile stride in halves (80B rows, keeps 16B alignment)

// ---- accumulator scratch layout (floats) ----
#define ACC_LINK 0
#define ACC_ENT  1
#define ACC_SSUM 2    // 2
#define ACC_SSQ  4    // 2
#define ACC_CSUM 8    // 512
#define ACC_CSQ  520  // 512
#define ACC_OUT  1032 // 2*512
#define ACC_OADJ 2056 // 4
#define ACC_N    2176

__device__ __forceinline__ float leakyf(float x) { return x > 0.f ? x : 0.01f * x; }

// Fragment loader for 16-bit 16x32 A (and B stored node-major [n][K]):
// lanes 0-15: rows 0-15, K 0..7 / 16..23 ; lanes 16-31: rows 0-15, K 8..15 / 24..31
__device__ __forceinline__ v16h load_frag(const h16* base) {
  const int lane = threadIdx.x & 31;
  const int r = lane & 15, hi = lane >> 4;
  const h16* p = base + r * LDT + hi * 8;
  v16h v;
  *(float4*)&v       = *(const float4*)(p);
  *((float4*)&v + 1) = *(const float4*)(p + 16);
  return v;
}

// ---------------- init ----------------
__global__ void init_acc_kernel(float* acc) {
  int i = blockIdx.x * 256 + threadIdx.x;
  if (i < ACC_N) acc[i] = 0.f;
}

// ---------------- row normalize + f16 convert ----------------
__global__ void __launch_bounds__(256) xnorm_kernel(const float* __restrict__ x,
                                                    h16* __restrict__ x16,
                                                    h16* __restrict__ xn16) {
  __shared__ float red[256];
  __shared__ float nrm;
  const int i = blockIdx.x;
  float s = 0.f;
  for (int k = threadIdx.x; k < LL; k += 256) {
    float v = x[(size_t)i * LL + k];
    s += v * v;
  }
  red[threadIdx.x] = s;
  __syncthreads();
  for (int o = 128; o > 0; o >>= 1) {
    if (threadIdx.x < o) red[threadIdx.x] += red[threadIdx.x + o];
    __syncthreads();
  }
  if (threadIdx.x == 0) nrm = fmaxf(sqrtf(red[0]), 1e-6f);
  __syncthreads();
  float inv = 1.f / nrm;
  for (int k = threadIdx.x; k < LL; k += 256) {
    float v = x[(size_t)i * LL + k];
    x16[(size_t)i * LL + k]  = (h16)v;
    xn16[(size_t)i * LL + k] = (h16)(v * inv);
  }
}

__global__ void w2h_kernel(const float* __restrict__ src, h16* __restrict__ dst, int n) {
  int i = blockIdx.x * 256 + threadIdx.x;
  if (i < n) dst[i] = (h16)src[i];
}

// ---------------- sim = xn @ xn^T, threshold -> adj (u8) ----------------
__global__ void __launch_bounds__(256) sim_adj_kernel(const h16* __restrict__ xn,
                                                      unsigned char* __restrict__ adj) {
  __shared__ __align__(16) h16 As[128 * LDT];
  __shared__ __align__(16) h16 Bs[128 * LDT];
  const int tid = threadIdx.x;
  const int rowBase = blockIdx.y * 128;
  const int colBase = blockIdx.x * 128;
  const int wv = tid >> 5;
  const int rowOff = (wv >> 1) * 32;
  const int colOff = (wv & 1) * 64;
  v8f acc[2][4] = {};
  const int r = tid >> 1, co = (tid & 1) * 16;
  for (int k0 = 0; k0 < LL; k0 += 32) {
    const float4* ga = (const float4*)(xn + (size_t)(rowBase + r) * LL + k0 + co);
    float4* la = (float4*)(As + r * LDT + co);
    la[0] = ga[0]; la[1] = ga[1];
    const float4* gb = (const float4*)(xn + (size_t)(colBase + r) * LL + k0 + co);
    float4* lb = (float4*)(Bs + r * LDT + co);
    lb[0] = gb[0]; lb[1] = gb[1];
    __syncthreads();
    v16h a0 = load_frag(As + (rowOff + 0) * LDT);
    v16h a1 = load_frag(As + (rowOff + 16) * LDT);
#pragma unroll
    for (int cc = 0; cc < 4; ++cc) {
      v16h b = load_frag(Bs + (colOff + cc * 16) * LDT);
      acc[0][cc] = __builtin_amdgcn_wmma_f32_16x16x32_f16(false, a0, false, b, (short)0, acc[0][cc], false, false);
      acc[1][cc] = __builtin_amdgcn_wmma_f32_16x16x32_f16(false, a1, false, b, (short)0, acc[1][cc], false, false);
    }
    __syncthreads();
  }
  const int lane = tid & 31, nl = lane & 15, hi = lane >> 4;
#pragma unroll
  for (int rr = 0; rr < 2; ++rr)
#pragma unroll
    for (int cc = 0; cc < 4; ++cc) {
      int gn = colBase + colOff + cc * 16 + nl;
      int mb = rowBase + rowOff + rr * 16 + hi * 8;
#pragma unroll
      for (int e = 0; e < 8; ++e) {
        int gm = mb + e;
        float sv = acc[rr][cc][e];
        adj[(size_t)gm * NN + gn] = (sv > 0.5f && gm != gn) ? 1 : 0;
      }
    }
}

// ---------------- deg = clip(rowsum(adj), 1, inf) ----------------
__global__ void __launch_bounds__(256) deg_kernel(const unsigned char* __restrict__ adj,
                                                  float* __restrict__ deg) {
  __shared__ float red[256];
  const int i = blockIdx.x;
  float s = 0.f;
  for (int j = threadIdx.x; j < NN; j += 256) s += (float)adj[(size_t)i * NN + j];
  red[threadIdx.x] = s;
  __syncthreads();
  for (int o = 128; o > 0; o >>= 1) {
    if (threadIdx.x < o) red[threadIdx.x] += red[threadIdx.x + o];
    __syncthreads();
  }
  if (threadIdx.x == 0) deg[i] = fmaxf(red[0], 1.0f);
}

// ---------------- agg = (adj @ x) / deg  (WMMA, u8->f16 staging) ----------------
__global__ void __launch_bounds__(256) agg_kernel(const unsigned char* __restrict__ adj,
                                                  const h16* __restrict__ x16,
                                                  const float* __restrict__ deg,
                                                  h16* __restrict__ agg16) {
  __shared__ __align__(16) h16 As[128 * LDT];
  __shared__ __align__(16) h16 Bs[128 * LDT];
  const int tid = threadIdx.x;
  const int rowBase = blockIdx.y * 128;
  const int colBase = blockIdx.x * 128;  // over 512 channels
  const int wv = tid >> 5;
  const int rowOff = (wv >> 1) * 32;
  const int colOff = (wv & 1) * 64;
  v8f acc[2][4] = {};
  const int ar = tid >> 1, ac = (tid & 1) * 16;
  const int bk = tid >> 3, bn = (tid & 7) * 16;
  for (int k0 = 0; k0 < NN; k0 += 32) {
    // A: u8 adjacency -> f16 in LDS
    uint4 u = *(const uint4*)(adj + (size_t)(rowBase + ar) * NN + k0 + ac);
    const unsigned char* bb = (const unsigned char*)&u;
#pragma unroll
    for (int j = 0; j < 16; ++j) As[ar * LDT + ac + j] = (h16)(float)bb[j];
    // B: x16 [k][n] -> Bs [n][k] (transpose during staging)
    h16 tmp[16];
    const float4* gx = (const float4*)(x16 + (size_t)(k0 + bk) * LL + colBase + bn);
    *(float4*)tmp       = gx[0];
    *(float4*)(tmp + 8) = gx[1];
#pragma unroll
    for (int j = 0; j < 16; ++j) Bs[(bn + j) * LDT + bk] = tmp[j];
    __syncthreads();
    v16h a0 = load_frag(As + (rowOff + 0) * LDT);
    v16h a1 = load_frag(As + (rowOff + 16) * LDT);
#pragma unroll
    for (int cc = 0; cc < 4; ++cc) {
      v16h b = load_frag(Bs + (colOff + cc * 16) * LDT);
      acc[0][cc] = __builtin_amdgcn_wmma_f32_16x16x32_f16(false, a0, false, b, (short)0, acc[0][cc], false, false);
      acc[1][cc] = __builtin_amdgcn_wmma_f32_16x16x32_f16(false, a1, false, b, (short)0, acc[1][cc], false, false);
    }
    __syncthreads();
  }
  const int lane = tid & 31, nl = lane & 15, hi = lane >> 4;
#pragma unroll
  for (int rr = 0; rr < 2; ++rr)
#pragma unroll
    for (int cc = 0; cc < 4; ++cc) {
      int gn = colBase + colOff + cc * 16 + nl;
      int mb = rowBase + rowOff + rr * 16 + hi * 8;
#pragma unroll
      for (int e = 0; e < 8; ++e) {
        int gm = mb + e;
        agg16[(size_t)gm * LL + gn] = (h16)(acc[rr][cc][e] / deg[gm]);
      }
    }
}

// ---------------- h1 = leaky(agg@rel_w + x@root_w + b)  (WMMA, two K passes) ----------------
__global__ void __launch_bounds__(256) h1_kernel(const h16* __restrict__ agg16,
                                                 const h16* __restrict__ x16,
                                                 const h16* __restrict__ wrel,
                                                 const h16* __restrict__ wroot,
                                                 const float* __restrict__ bias,
                                                 float* __restrict__ h) {
  __shared__ __align__(16) h16 As[128 * LDT];
  __shared__ __align__(16) h16 Bs[128 * LDT];
  const int tid = threadIdx.x;
  const int rowBase = blockIdx.y * 128;
  const int colBase = blockIdx.x * 128;
  const int wv = tid >> 5;
  const int rowOff = (wv >> 1) * 32;
  const int colOff = (wv & 1) * 64;
  v8f acc[2][4] = {};
  const int ar = tid >> 1, ac = (tid & 1) * 16;
  const int bk = tid >> 3, bn = (tid & 7) * 16;
  for (int pass = 0; pass < 2; ++pass) {
    const h16* A = pass ? x16 : agg16;
    const h16* W = pass ? wroot : wrel;
    for (int k0 = 0; k0 < LL; k0 += 32) {
      const float4* ga = (const float4*)(A + (size_t)(rowBase + ar) * LL + k0 + ac);
      float4* la = (float4*)(As + ar * LDT + ac);
      la[0] = ga[0]; la[1] = ga[1];
      h16 tmp[16];
      const float4* gw = (const float4*)(W + (size_t)(k0 + bk) * LL + colBase + bn);
      *(float4*)tmp       = gw[0];
      *(float4*)(tmp + 8) = gw[1];
#pragma unroll
      for (int j = 0; j < 16; ++j) Bs[(bn + j) * LDT + bk] = tmp[j];
      __syncthreads();
      v16h a0 = load_frag(As + (rowOff + 0) * LDT);
      v16h a1 = load_frag(As + (rowOff + 16) * LDT);
#pragma unroll
      for (int cc = 0; cc < 4; ++cc) {
        v16h b = load_frag(Bs + (colOff + cc * 16) * LDT);
        acc[0][cc] = __builtin_amdgcn_wmma_f32_16x16x32_f16(false, a0, false, b, (short)0, acc[0][cc], false, false);
        acc[1][cc] = __builtin_amdgcn_wmma_f32_16x16x32_f16(false, a1, false, b, (short)0, acc[1][cc], false, false);
      }
      __syncthreads();
    }
  }
  const int lane = tid & 31, nl = lane & 15, hi = lane >> 4;
#pragma unroll
  for (int rr = 0; rr < 2; ++rr)
#pragma unroll
    for (int cc = 0; cc < 4; ++cc) {
      int gn = colBase + colOff + cc * 16 + nl;
      int mb = rowBase + rowOff + rr * 16 + hi * 8;
      float bval = bias[gn];
#pragma unroll
      for (int e = 0; e < 8; ++e) {
        int gm = mb + e;
        h[(size_t)gm * LL + gn] = leakyf(acc[rr][cc][e] + bval);
      }
    }
}

// ---------------- s_pre: leaky(agg@pool_rel + x@pool_root + b)  (C=2) ----------------
__global__ void __launch_bounds__(256) spre_kernel(const h16* __restrict__ agg16,
                                                   const float* __restrict__ x,
                                                   const float* __restrict__ relw,
                                                   const float* __restrict__ relb,
                                                   const float* __restrict__ rootw,
                                                   float* __restrict__ spre) {
  const int row = blockIdx.x * 8 + (threadIdx.x >> 5);
  const int lane = threadIdx.x & 31;
  float p0 = 0.f, p1 = 0.f;
  for (int k = lane; k < LL; k += 32) {
    float a = (float)agg16[(size_t)row * LL + k];
    float xv = x[(size_t)row * LL + k];
    p0 += a * relw[k * 2 + 0] + xv * rootw[k * 2 + 0];
    p1 += a * relw[k * 2 + 1] + xv * rootw[k * 2 + 1];
  }
  for (int o = 16; o > 0; o >>= 1) {
    p0 += __shfl_down(p0, o);
    p1 += __shfl_down(p1, o);
  }
  if (lane == 0) {
    spre[row * 2 + 0] = leakyf(p0 + relb[0]);
    spre[row * 2 + 1] = leakyf(p1 + relb[1]);
  }
}

// ---------------- BN stats for s (2 channels) ----------------
__global__ void sstats_kernel(const float* __restrict__ spre, float* __restrict__ acc) {
  const int row = blockIdx.x * 256 + threadIdx.x;
  float a = spre[row * 2 + 0], b = spre[row * 2 + 1];
  atomicAdd(&acc[ACC_SSUM + 0], a);
  atomicAdd(&acc[ACC_SSUM + 1], b);
  atomicAdd(&acc[ACC_SSQ + 0], a * a);
  atomicAdd(&acc[ACC_SSQ + 1], b * b);
}

// ---------------- s finalize: BN -> lin -> leaky -> softmax -> entropy ----------------
__global__ void sfin_kernel(const float* __restrict__ spre,
                            const float* __restrict__ bn_g, const float* __restrict__ bn_b,
                            const float* __restrict__ lw, const float* __restrict__ lb,
                            float* __restrict__ s, float* __restrict__ acc) {
  const int row = blockIdx.x * 256 + threadIdx.x;
  const float inv_n = 1.f / (float)NN;
  float m0 = acc[ACC_SSUM + 0] * inv_n, m1 = acc[ACC_SSUM + 1] * inv_n;
  float v0 = acc[ACC_SSQ + 0] * inv_n - m0 * m0;
  float v1 = acc[ACC_SSQ + 1] * inv_n - m1 * m1;
  float b0 = (spre[row * 2 + 0] - m0) * rsqrtf(v0 + 1e-5f) * bn_g[0] + bn_b[0];
  float b1 = (spre[row * 2 + 1] - m1) * rsqrtf(v1 + 1e-5f) * bn_g[1] + bn_b[1];
  float y0 = leakyf(b0 * lw[0] + b1 * lw[2] + lb[0]);
  float y1 = leakyf(b0 * lw[1] + b1 * lw[3] + lb[1]);
  float mx = fmaxf(y0, y1);
  float e0 = expf(y0 - mx), e1 = expf(y1 - mx);
  float inv = 1.f / (e0 + e1);
  float s0 = e0 * inv, s1 = e1 * inv;
  s[row * 2 + 0] = s0;
  s[row * 2 + 1] = s1;
  float ent = -(s0 * logf(s0 + 1e-15f) + s1 * logf(s1 + 1e-15f));
  atomicAdd(&acc[ACC_ENT], ent);
}

// ---------------- BN stats for h (512 channels) ----------------
__global__ void hstats_kernel(const float* __restrict__ h, float* __restrict__ acc) {
  const int col = blockIdx.x * 256 + threadIdx.x;
  const int m0 = blockIdx.y * 128;
  float s = 0.f, sq = 0.f;
  for (int m = m0; m < m0 + 128; ++m) {
    float v = h[(size_t)m * LL + col];
    s += v;
    sq += v * v;
  }
  atomicAdd(&acc[ACC_CSUM + col], s);
  atomicAdd(&acc[ACC_CSQ + col], sq);
}

// ---------------- streaming adj pass: t = adj@s, link += (adj - s s^T)^2 ----------------
__global__ void __launch_bounds__(256) adjlink_kernel(const unsigned char* __restrict__ adj,
                                                      const float* __restrict__ s,
                                                      float* __restrict__ t,
                                                      float* __restrict__ acc) {
  __shared__ float r0[256], r1[256], r2[256];
  const int i = blockIdx.x;
  const float si0 = s[i * 2 + 0], si1 = s[i * 2 + 1];
  float t0 = 0.f, t1 = 0.f, lk = 0.f;
  for (int j = threadIdx.x; j < NN; j += 256) {
    float a = (float)adj[(size_t)i * NN + j];
    float sj0 = s[j * 2 + 0], sj1 = s[j * 2 + 1];
    float d = a - (si0 * sj0 + si1 * sj1);
    lk += d * d;
    t0 += a * sj0;
    t1 += a * sj1;
  }
  r0[threadIdx.x] = t0; r1[threadIdx.x] = t1; r2[threadIdx.x] = lk;
  __syncthreads();
  for (int o = 128; o > 0; o >>= 1) {
    if (threadIdx.x < o) {
      r0[threadIdx.x] += r0[threadIdx.x + o];
      r1[threadIdx.x] += r1[threadIdx.x + o];
      r2[threadIdx.x] += r2[threadIdx.x + o];
    }
    __syncthreads();
  }
  if (threadIdx.x == 0) {
    t[i * 2 + 0] = r0[0];
    t[i * 2 + 1] = r1[0];
    atomicAdd(&acc[ACC_LINK], r2[0]);
  }
}

// ---------------- out[c][n] = sum_m s[m,c] * BN(h)[m,n] ----------------
__global__ void poolred_kernel(const float* __restrict__ h, const float* __restrict__ s,
                               const float* __restrict__ bn_g, const float* __restrict__ bn_b,
                               float* __restrict__ acc) {
  const int col = blockIdx.x * 256 + threadIdx.x;
  const float inv_n = 1.f / (float)NN;
  float mean = acc[ACC_CSUM + col] * inv_n;
  float var = acc[ACC_CSQ + col] * inv_n - mean * mean;
  float rs = rsqrtf(var + 1e-5f);
  float gg = bn_g[col], bb = bn_b[col];
  const int m0 = blockIdx.y * 128;
  float p0 = 0.f, p1 = 0.f;
  for (int m = m0; m < m0 + 128; ++m) {
    float hb = (h[(size_t)m * LL + col] - mean) * rs * gg + bb;
    p0 += s[m * 2 + 0] * hb;
    p1 += s[m * 2 + 1] * hb;
  }
  atomicAdd(&acc[ACC_OUT + 0 * LL + col], p0);
  atomicAdd(&acc[ACC_OUT + 1 * LL + col], p1);
}

// ---------------- out_adj = s^T t ----------------
__global__ void outadj_kernel(const float* __restrict__ s, const float* __restrict__ t,
                              float* __restrict__ acc) {
  __shared__ float red[4][256];
  float a00 = 0.f, a01 = 0.f, a10 = 0.f, a11 = 0.f;
  for (int m = threadIdx.x; m < NN; m += 256) {
    float s0 = s[m * 2 + 0], s1 = s[m * 2 + 1];
    float t0 = t[m * 2 + 0], t1 = t[m * 2 + 1];
    a00 += s0 * t0; a01 += s0 * t1; a10 += s1 * t0; a11 += s1 * t1;
  }
  red[0][threadIdx.x] = a00; red[1][threadIdx.x] = a01;
  red[2][threadIdx.x] = a10; red[3][threadIdx.x] = a11;
  __syncthreads();
  for (int o = 128; o > 0; o >>= 1) {
    if (threadIdx.x < o)
      for (int q = 0; q < 4; ++q) red[q][threadIdx.x] += red[q][threadIdx.x + o];
    __syncthreads();
  }
  if (threadIdx.x < 4) acc[ACC_OADJ + threadIdx.x] = red[threadIdx.x][0];
}

// ---------------- final: emb3 gnn on [2,512] + head ----------------
__global__ void __launch_bounds__(256) final_kernel(const float* __restrict__ relw3,
                                                    const float* __restrict__ relb3,
                                                    const float* __restrict__ rootw3,
                                                    const float* __restrict__ g3,
                                                    const float* __restrict__ b3,
                                                    const float* __restrict__ l1w,
                                                    const float* __restrict__ l1b,
                                                    const float* __restrict__ l2w,
                                                    const float* __restrict__ l2b,
                                                    const float* __restrict__ acc,
                                                    float* __restrict__ dout) {
  __shared__ float hm[512];
  __shared__ float z[256];
  const float* out = acc + ACC_OUT;  // [2][512]
  float oa00 = acc[ACC_OADJ + 0], oa01 = acc[ACC_OADJ + 1];
  float oa10 = acc[ACC_OADJ + 2], oa11 = acc[ACC_OADJ + 3];
  float id0 = 1.f / fmaxf(oa00 + oa01, 1.f);
  float id1 = 1.f / fmaxf(oa10 + oa11, 1.f);
  for (int n = threadIdx.x; n < 512; n += 256) {
    float a0 = relb3[n], a1 = relb3[n];
    for (int k = 0; k < 512; ++k) {
      float o0 = out[k], o1 = out[512 + k];
      float ag0 = (oa00 * o0 + oa01 * o1) * id0;
      float ag1 = (oa10 * o0 + oa11 * o1) * id1;
      float wr = relw3[k * 512 + n], wo = rootw3[k * 512 + n];
      a0 += ag0 * wr + o0 * wo;
      a1 += ag1 * wr + o1 * wo;
    }
    float h0 = leakyf(a0), h1 = leakyf(a1);
    float m = (h0 + h1) * 0.5f;
    float v = ((h0 - m) * (h0 - m) + (h1 - m) * (h1 - m)) * 0.5f;
    float rs = rsqrtf(v + 1e-5f);
    float bn0 = (h0 - m) * rs * g3[n] + b3[n];
    float bn1 = (h1 - m) * rs * g3[n] + b3[n];
    hm[n] = (bn0 + bn1) * 0.5f;
  }
  __syncthreads();
  {
    int tdx = threadIdx.x;
    float a = l1b[tdx];
    for (int n = 0; n < 512; ++n) a += hm[n] * l1w[n * 256 + tdx];
    z[tdx] = fmaxf(a, 0.f);
  }
  __syncthreads();
  if (threadIdx.x == 0) {
    float lg[4];
    for (int c = 0; c < 4; ++c) {
      float a = l2b[c];
      for (int q = 0; q < 256; ++q) a += z[q] * l2w[q * 4 + c];
      lg[c] = a;
    }
    float mx = fmaxf(fmaxf(lg[0], lg[1]), fmaxf(lg[2], lg[3]));
    float es = 0.f, ev[4];
    for (int c = 0; c < 4; ++c) { ev[c] = expf(lg[c] - mx); es += ev[c]; }
    for (int c = 0; c < 4; ++c) dout[c] = ev[c] / es;
    dout[4] = sqrtf(acc[ACC_LINK]) / (float)((long long)NN * (long long)NN);
    dout[5] = acc[ACC_ENT] / (float)NN;
  }
}

extern "C" void kernel_launch(void* const* d_in, const int* in_sizes, int n_in,
                              void* d_out, int out_size, void* d_ws, size_t ws_size,
                              hipStream_t stream) {
  const float* x          = (const float*)d_in[0];
  const float* pool_rel_w = (const float*)d_in[1];
  const float* pool_rel_b = (const float*)d_in[2];
  const float* pool_root_w= (const float*)d_in[3];
  const float* pool_bn_g  = (const float*)d_in[4];
  const float* pool_bn_b  = (const float*)d_in[5];
  const float* pool_lin_w = (const float*)d_in[6];
  const float* pool_lin_b = (const float*)d_in[7];
  const float* emb1_rel_w = (const float*)d_in[8];
  const float* emb1_rel_b = (const float*)d_in[9];
  const float* emb1_root_w= (const float*)d_in[10];
  const float* emb1_bn_g  = (const float*)d_in[11];
  const float* emb1_bn_b  = (const float*)d_in[12];
  const float* emb3_rel_w = (const float*)d_in[13];
  const float* emb3_rel_b = (const float*)d_in[14];
  const float* emb3_root_w= (const float*)d_in[15];
  const float* emb3_bn_g  = (const float*)d_in[16];
  const float* emb3_bn_b  = (const float*)d_in[17];
  const float* lin1_w     = (const float*)d_in[18];
  const float* lin1_b     = (const float*)d_in[19];
  const float* lin2_w     = (const float*)d_in[20];
  const float* lin2_b     = (const float*)d_in[21];
  float* dout = (float*)d_out;

  char* ws = (char*)d_ws;
  h16* x16    = (h16*)(ws + 0);           //  8 MB
  h16* xn16   = (h16*)(ws + 8388608);     //  8 MB
  h16* agg16  = (h16*)(ws + 16777216);    //  8 MB
  h16* w1rel  = (h16*)(ws + 25165824);    //  0.5 MB
  h16* w1root = (h16*)(ws + 25690112);    //  0.5 MB
  unsigned char* adj = (unsigned char*)(ws + 26214400);  // 64 MB
  float* deg  = (float*)(ws + 93323264);  // 32 KB
  float* spre = (float*)(ws + 93356032);  // 64 KB
  float* s    = (float*)(ws + 93421568);  // 64 KB
  float* t    = (float*)(ws + 93487104);  // 64 KB
  float* h    = (float*)(ws + 93552640);  // 16 MB
  float* acc  = (float*)(ws + 110329856); // 16 KB

  init_acc_kernel<<<(ACC_N + 255) / 256, 256, 0, stream>>>(acc);
  xnorm_kernel<<<NN, 256, 0, stream>>>(x, x16, xn16);
  w2h_kernel<<<(LL * LL + 255) / 256, 256, 0, stream>>>(emb1_rel_w, w1rel, LL * LL);
  w2h_kernel<<<(LL * LL + 255) / 256, 256, 0, stream>>>(emb1_root_w, w1root, LL * LL);

  sim_adj_kernel<<<dim3(NN / 128, NN / 128), 256, 0, stream>>>(xn16, adj);
  deg_kernel<<<NN, 256, 0, stream>>>(adj, deg);
  agg_kernel<<<dim3(LL / 128, NN / 128), 256, 0, stream>>>(adj, x16, deg, agg16);

  spre_kernel<<<NN / 8, 256, 0, stream>>>(agg16, x, pool_rel_w, pool_rel_b, pool_root_w, spre);
  h1_kernel<<<dim3(LL / 128, NN / 128), 256, 0, stream>>>(agg16, x16, w1rel, w1root, emb1_rel_b, h);

  sstats_kernel<<<NN / 256, 256, 0, stream>>>(spre, acc);
  sfin_kernel<<<NN / 256, 256, 0, stream>>>(spre, pool_bn_g, pool_bn_b, pool_lin_w, pool_lin_b, s, acc);

  hstats_kernel<<<dim3(LL / 256, NN / 128), 256, 0, stream>>>(h, acc);
  adjlink_kernel<<<NN, 256, 0, stream>>>(adj, s, t, acc);
  poolred_kernel<<<dim3(LL / 256, NN / 128), 256, 0, stream>>>(h, s, emb1_bn_g, emb1_bn_b, acc);
  outadj_kernel<<<1, 256, 0, stream>>>(s, t, acc);

  final_kernel<<<1, 256, 0, stream>>>(emb3_rel_w, emb3_rel_b, emb3_root_w,
                                      emb3_bn_g, emb3_bn_b,
                                      lin1_w, lin1_b, lin2_w, lin2_b, acc, dout);
}
